// DTWLosslabels_57131654971997
// MI455X (gfx1250) — compile-verified, matrix operand order checked by reading, not run on hardware
//
#include <hip/hip_runtime.h>
#include <hip/hip_bf16.h>

// --- types for WMMA fragments (CDNA5 gfx1250, wave32) ---
typedef __attribute__((ext_vector_type(16))) __bf16 v16bf;
typedef __attribute__((ext_vector_type(8)))  __bf16 v8bf;
typedef __attribute__((ext_vector_type(8)))  float  v8f;

static constexpr int   BATCH = 64;
static constexpr int   T     = 512;
static constexpr int   DIM   = 64;
static constexpr int   SROWS = 32;        // strip height = full wave32 anti-diagonal
static constexpr int   NSTRIP = T / SROWS;          // 16
static constexpr int   ROWSTRIDE = 514;   // pad: bank = (2r + j) % 64 distinct for r=0..31
static constexpr float BIGV  = 1e8f;

__device__ __forceinline__ float softmin3(float a, float b, float c) {
  // -log(e^-a + e^-b + e^-c), min-shifted; exp(m - BIG) flushes to 0
  float m = fminf(a, fminf(b, c));
  float s = __expf(m - a) + __expf(m - b) + __expf(m - c);
  return m - __logf(s);
}

// ---------------------------------------------------------------------------
// Kernel 1: L2-normalize rows of TGT/OTH/X, write bf16 into workspace.
// Layout in ws: [0]=TGT_bf, [1]=OTH_bf, [2]=X_bf, each B*T*DIM bf16.
// ---------------------------------------------------------------------------
__global__ void norm_rows(const float* __restrict__ tgt,
                          const float* __restrict__ oth,
                          const float* __restrict__ x,
                          unsigned short* __restrict__ out_u) {
  int row = blockIdx.x * blockDim.x + threadIdx.x;
  const int total = 3 * BATCH * T;
  if (row >= total) return;
  int which = row / (BATCH * T);
  int r = row - which * (BATCH * T);
  const float* src = (which == 0 ? tgt : (which == 1 ? oth : x)) + (size_t)r * DIM;
  __bf16* dst = (__bf16*)out_u + (size_t)which * (BATCH * T * DIM) + (size_t)r * DIM;

  float v[DIM];
  float ss = 0.0f;
#pragma unroll
  for (int k = 0; k < DIM; k += 4) {
    float4 f = *(const float4*)(src + k);
    v[k] = f.x; v[k + 1] = f.y; v[k + 2] = f.z; v[k + 3] = f.w;
    ss += f.x * f.x + f.y * f.y + f.z * f.z + f.w * f.w;
  }
  float inv = 1.0f / fmaxf(sqrtf(ss), 1e-8f);
#pragma unroll
  for (int k = 0; k < DIM; ++k) dst[k] = (__bf16)(v[k] * inv);
}

// ---------------------------------------------------------------------------
// Distance tile: one wave computes a 16x16 tile of Dist = 1 - Xn * Yn^T via
// two v_wmma_f32_16x16x32_bf16 (K = 64 = 2 x 32), writing fp32 into LDS.
// 16-bit A-matrix 16x32 layout: lanes 0-15 hold M=0..15 with K in
// {kb..kb+7, kb+16..kb+23}, lanes 16-31 hold the K+8 halves (kb = 8).
// B (32x16, column-major source) mirrors this with N instead of M.
// Strip tile index t in [0,64): row-half = t>>5, column tile = t&31.
// ---------------------------------------------------------------------------
__device__ __forceinline__ void fill_tile(const __bf16* __restrict__ Xr,
                                          const __bf16* __restrict__ Yr,
                                          float* __restrict__ dst,
                                          int stripBase, int t, int lane) {
  const int rowHalf = t >> 5;            // 0 or 1 (which 16-row half of the strip)
  const int j0      = (t & 31) * 16;
  const int rsel    = lane & 15;
  const int kb      = (lane < 16) ? 0 : 8;

  const __bf16* ax = Xr + (size_t)(stripBase + rowHalf * 16 + rsel) * DIM + kb;
  const __bf16* by = Yr + (size_t)(j0 + rsel) * DIM + kb;

  __builtin_prefetch(by + DIM * 16, 0, 0);  // hint: next tile's Y rows (global_prefetch_b8)

  v8bf a0 = *(const v8bf*)(ax +  0);
  v8bf a1 = *(const v8bf*)(ax + 16);
  v8bf a2 = *(const v8bf*)(ax + 32);
  v8bf a3 = *(const v8bf*)(ax + 48);
  v8bf b0 = *(const v8bf*)(by +  0);
  v8bf b1 = *(const v8bf*)(by + 16);
  v8bf b2 = *(const v8bf*)(by + 32);
  v8bf b3 = *(const v8bf*)(by + 48);

  v16bf A0 = __builtin_shufflevector(a0, a1, 0,1,2,3,4,5,6,7,8,9,10,11,12,13,14,15);
  v16bf A1 = __builtin_shufflevector(a2, a3, 0,1,2,3,4,5,6,7,8,9,10,11,12,13,14,15);
  v16bf B0 = __builtin_shufflevector(b0, b1, 0,1,2,3,4,5,6,7,8,9,10,11,12,13,14,15);
  v16bf B1 = __builtin_shufflevector(b2, b3, 0,1,2,3,4,5,6,7,8,9,10,11,12,13,14,15);

  v8f c = {};
  c = __builtin_amdgcn_wmma_f32_16x16x32_bf16(false, A0, false, B0, (short)0, c, false, false);
  c = __builtin_amdgcn_wmma_f32_16x16x32_bf16(false, A1, false, B1, (short)0, c, false, false);

  const int mbase = rowHalf * 16 + ((lane < 16) ? 0 : 8);
#pragma unroll
  for (int g = 0; g < 8; ++g)
    dst[(mbase + g) * ROWSTRIDE + j0 + rsel] = 1.0f - c[g];
}

// ---------------------------------------------------------------------------
// Kernel 2: one workgroup per (pair, batch) matrix. 256 threads = 8 waves.
// Wave 0 runs the soft-DTW skewed wavefront (all 32 lanes = 32-row strips);
// waves 1..7 produce the next strip's distances with WMMA (double-buffered
// LDS). 16 strips x 543 serial steps — half of the 16-row-strip variant.
// ---------------------------------------------------------------------------
__global__ void sdtw_kernel(const unsigned short* __restrict__ xb_u,
                            float* __restrict__ results) {
  extern __shared__ float smem[];
  float* dsbuf0  = smem;                               // SROWS x ROWSTRIDE
  float* dsbuf1  = smem + SROWS * ROWSTRIDE;
  float* prevRow = smem + 2 * SROWS * ROWSTRIDE;       // 513 floats: R[i-1][-1..511]

  const __bf16* xb = (const __bf16*)xb_u;
  const size_t AN = (size_t)BATCH * T * DIM;
  const __bf16* tgt = xb;
  const __bf16* oth = xb + AN;
  const __bf16* xx  = xb + 2 * AN;

  const int m = blockIdx.x;     // 0..255
  const int p = m >> 6;         // 0:(OTH,X) 1:(TGT,X) 2:(OTH,OTH) 3:(TGT,TGT)
  const int b = m & 63;
  const __bf16* Xr; const __bf16* Yr;
  if      (p == 0) { Xr = oth; Yr = xx;  }
  else if (p == 1) { Xr = tgt; Yr = xx;  }
  else if (p == 2) { Xr = oth; Yr = oth; }
  else             { Xr = tgt; Yr = tgt; }
  Xr += (size_t)b * T * DIM;
  Yr += (size_t)b * T * DIM;

  const int tid  = threadIdx.x;
  const int wv   = tid >> 5;
  const int lane = tid & 31;

  for (int j = tid; j <= T; j += blockDim.x) prevRow[j] = (j == 0) ? 0.0f : BIGV;

  // prefill strip 0 distances (all 8 waves, 8 of 64 tiles each)
  for (int t = wv; t < 64; t += 8) fill_tile(Xr, Yr, dsbuf0, 0, t, lane);
  __syncthreads();

  float finalR = 0.0f;
  for (int s = 0; s < NSTRIP; ++s) {
    float* cur = (s & 1) ? dsbuf1 : dsbuf0;
    float* nxt = (s & 1) ? dsbuf0 : dsbuf1;

    if (wv != 0) {
      if (s + 1 < NSTRIP)
        for (int t = wv - 1; t < 64; t += 7)
          fill_tile(Xr, Yr, nxt, (s + 1) * SROWS, t, lane);
    } else {
      // --- sequential skewed wavefront over 32 rows x 512 cols ---
      const int r = lane;               // lane r carries row s*32 + r
      const float prev0 = prevRow[0];   // R[i0-1][-1]  (0 for strip 0, BIG after)
      float left = BIGV;                // R[i][j-1]
      float diag = (r == 0) ? prev0 : BIGV;  // R[i-1][j-1] at j=0
      float myR  = BIGV;
      for (int t = 0; t < T + (SROWS - 1); ++t) {
        float upsh = __shfl_up(myR, 1, 32);          // lane r-1's R[i-1][j]
        int j = t - r;
        bool act = (j >= 0) & (j < T);
        float up = (r == 0) ? ((j >= 0 && j < T) ? prevRow[j + 1] : BIGV) : upsh;
        if (act) {
          float d   = cur[r * ROWSTRIDE + j];
          float val = d + softmin3(up, left, diag);
          diag = up; left = val; myR = val;
          if (r == SROWS - 1) {
            prevRow[j + 1] = val;        // trails lane0's reads by 31 steps
            if (s == NSTRIP - 1 && j == T - 1) finalR = val;  // R[511][511]
          }
        }
      }
      if (r == 0) prevRow[0] = BIGV;     // R[i-1][-1] = BIG for later strips
    }
    __syncthreads();
  }
  if (tid == SROWS - 1) results[m] = finalR;
}

// ---------------------------------------------------------------------------
// Kernel 3: combine the 4 sdtw terms per batch element, MSE vs labels.
// diff_b = S0 - S1 - 0.5*S2 + 0.5*S3   (the sdtw(X,X) terms cancel)
// ---------------------------------------------------------------------------
__global__ void finalize_loss(const float* __restrict__ results,
                              const float* __restrict__ labels,
                              float* __restrict__ out) {
  __shared__ float red[BATCH];
  int b = threadIdx.x;
  float diff = results[b] - results[64 + b] - 0.5f * results[128 + b] + 0.5f * results[192 + b];
  float e = diff - labels[b];
  red[b] = e * e;
  __syncthreads();
  if (b == 0) {
    float s = 0.0f;
    for (int i = 0; i < BATCH; ++i) s += red[i];
    out[0] = s / (float)BATCH;
  }
}

// ---------------------------------------------------------------------------
extern "C" void kernel_launch(void* const* d_in, const int* in_sizes, int n_in,
                              void* d_out, int out_size, void* d_ws, size_t ws_size,
                              hipStream_t stream) {
  (void)in_sizes; (void)n_in; (void)out_size; (void)ws_size;
  const float* tgt    = (const float*)d_in[0];
  const float* oth    = (const float*)d_in[1];
  const float* x      = (const float*)d_in[2];
  const float* labels = (const float*)d_in[3];

  unsigned short* xb = (unsigned short*)d_ws;                       // 3 * B*T*DIM bf16
  size_t bfBytes = (size_t)3 * BATCH * T * DIM * sizeof(unsigned short);
  float* results = (float*)((char*)d_ws + ((bfBytes + 255) & ~(size_t)255));  // 256 floats

  const int rows = 3 * BATCH * T;
  hipLaunchKernelGGL(norm_rows, dim3((rows + 255) / 256), dim3(256), 0, stream,
                     tgt, oth, x, xb);

  const size_t ldsBytes = (size_t)(2 * SROWS * ROWSTRIDE + 513 + 3) * sizeof(float);
  hipLaunchKernelGGL(sdtw_kernel, dim3(256), dim3(256), ldsBytes, stream,
                     xb, results);

  hipLaunchKernelGGL(finalize_loss, dim3(1), dim3(BATCH), 0, stream,
                     results, labels, (float*)d_out);
}
